// MambaDecoder_50740743635362
// MI455X (gfx1250) — matrix operand order repaired and checked by reading, not compiled, not run on hardware
//
#include <hip/hip_runtime.h>

// ---------------- problem constants (match reference) ----------------
#define BB      4
#define TT_IN   2048
#define D_IN    513
#define NC      512
#define T_OUT   509
#define PATCH   14
#define STRIDE_P 4
#define D_MODEL 512
#define PROJ    4096
#define NLAYER  5
#define DSTATE  64
#define DCONV   4
#define HEADDIM 64
#define DI      1024          // 2*D_MODEL
#define NH      16            // DI/HEADDIM
#define CONVD   1152          // DI + 2*DSTATE
#define DINPROJ 2192          // 2*DI + 2*DSTATE + NH
#define NMAMBA  10
#define N_CLASSES 41
#define ROWS    (BB*T_OUT)    // 2036
#define EPSV    1e-5f

typedef __bf16 bf16_t;
typedef __attribute__((ext_vector_type(16))) __bf16 v16bf;
typedef __attribute__((ext_vector_type(8)))  __bf16 bf16x8;    // 16B aligned (LDS)
typedef __bf16 bf16x8u __attribute__((ext_vector_type(8), aligned(4)));  // global, 4B aligned
typedef float  float4u __attribute__((ext_vector_type(4), aligned(4)));  // global, 4B aligned
typedef __attribute__((ext_vector_type(8)))  float  v8f;

static __device__ __forceinline__ v16bf mk16(bf16x8 lo, bf16x8 hi) {
  v16bf r;
#pragma unroll
  for (int i = 0; i < 8; ++i) { r[i] = lo[i]; r[i + 8] = hi[i]; }
  return r;
}
static __device__ __forceinline__ float softsign(float v) { return v / (1.f + fabsf(v)); }
static __device__ __forceinline__ float silu(float v)     { return v / (1.f + expf(-v)); }

// =====================================================================
// Generic bf16 WMMA GEMM:  C[M,N](f32) = epi( A[M,K](f32) * B[N,K](bf16)^T + bias )
// B is row-major [N][K] (native layout of all torch/jax weights: out x in).
// Block tile 128x128, 8 waves; each wave owns 32 rows x 64 cols = 2x4 WMMA tiles.
// ACT: 0 none, 1 softsign.  ACCUM: 0 store, 1 += with optional time-reverse remap.
// =====================================================================
#define BM 128
#define BN 128
#define BK 32

template <int ACT, int ACCUM>
__global__ __launch_bounds__(256) void wmma_gemm(
    const float* __restrict__ A, long long aBatch, int lda,
    const bf16_t* __restrict__ Bw, long long bBatch, int ldbK,
    const float* __restrict__ bias, long long biasBatch,
    float* __restrict__ C, long long cBatch, int ldc,
    int M, int N, int K, int remapT, int rev)
{
  __shared__ __align__(16) bf16_t As[BM][40];   // row stride 80B -> 16B-aligned chunks
  __shared__ __align__(16) bf16_t Bs[BN][40];   // [n][k]

  const int bz = blockIdx.z;
  A  += (long long)bz * aBatch;
  Bw += (long long)bz * bBatch;
  C  += (long long)bz * cBatch;
  const float* biasp = bias ? bias + (long long)bz * biasBatch : nullptr;

  const int m0 = blockIdx.y * BM;
  const int n0 = blockIdx.x * BN;
  const int tid  = threadIdx.x;
  const int lane = tid & 31;
  const int wave = tid >> 5;
  const int wm = wave >> 1;      // 0..3 : 32-row group
  const int wn = wave & 1;       // 0..1 : 64-col group
  const int half = lane >> 4;
  const int l16  = lane & 15;

  // staging coords: each thread stages half a row (16 elems) of A and of B
  const int ar = tid >> 1, ac = (tid & 1) * 16;
  const int bn = tid >> 1, bk = (tid & 1) * 16;

  const bool fullM = (m0 + BM <= M);
  const bool fullN = (n0 + BN <= N);

  v8f acc[2][4] = {};

  for (int k0 = 0; k0 < K; k0 += BK) {
    const bool fullK = (k0 + BK <= K);

    // ---------- stage A (f32 -> bf16) ----------
    float fa[16];
    if (fullM && fullK) {
      const float* ap = A + (long long)(m0 + ar) * lda + k0 + ac;
#pragma unroll
      for (int q = 0; q < 4; ++q) {
        float4u v = *(const float4u*)(ap + 4 * q);
#pragma unroll
        for (int e = 0; e < 4; ++e) fa[4 * q + e] = v[e];
      }
      bf16x8 p0, p1;
#pragma unroll
      for (int i = 0; i < 8; ++i) { p0[i] = (bf16_t)fa[i]; p1[i] = (bf16_t)fa[i + 8]; }
      *(bf16x8*)&As[ar][ac]     = p0;
      *(bf16x8*)&As[ar][ac + 8] = p1;
    } else {
      int gm = m0 + ar;
      bool rowOk = gm < M;
      long long rbase = (long long)(rowOk ? gm : 0) * lda;
#pragma unroll
      for (int i = 0; i < 16; ++i) {           // clamped loads: batch, no exec branches
        int gk = k0 + ac + i;
        fa[i] = A[rbase + (gk < K ? gk : 0)];
      }
      bf16x8 p0, p1;
#pragma unroll
      for (int i = 0; i < 16; ++i) {
        int gk = k0 + ac + i;
        bf16_t v = (rowOk && gk < K) ? (bf16_t)fa[i] : (bf16_t)0.f;
        if (i < 8) p0[i] = v; else p1[i - 8] = v;
      }
      *(bf16x8*)&As[ar][ac]     = p0;
      *(bf16x8*)&As[ar][ac + 8] = p1;
    }

    // ---------- stage B (bf16, already [N][K]) ----------
    if (fullN && fullK) {
      const bf16_t* bp = Bw + (long long)(n0 + bn) * ldbK + k0 + bk;
      bf16x8u b0 = *(const bf16x8u*)bp;
      bf16x8u b1 = *(const bf16x8u*)(bp + 8);
      bf16x8 q0, q1;
#pragma unroll
      for (int i = 0; i < 8; ++i) { q0[i] = b0[i]; q1[i] = b1[i]; }
      *(bf16x8*)&Bs[bn][bk]     = q0;
      *(bf16x8*)&Bs[bn][bk + 8] = q1;
    } else {
      int gn = n0 + bn;
      bool nOk = gn < N;
      long long nbase = (long long)(nOk ? gn : 0) * ldbK;
      bf16_t vb[16];
#pragma unroll
      for (int i = 0; i < 16; ++i) {
        int gk = k0 + bk + i;
        vb[i] = Bw[nbase + (gk < K ? gk : 0)];
      }
      bf16x8 q0, q1;
#pragma unroll
      for (int i = 0; i < 16; ++i) {
        int gk = k0 + bk + i;
        bf16_t v = (nOk && gk < K) ? vb[i] : (bf16_t)0.f;
        if (i < 8) q0[i] = v; else q1[i - 8] = v;
      }
      *(bf16x8*)&Bs[bn][bk]     = q0;
      *(bf16x8*)&Bs[bn][bk + 8] = q1;
    }
    __syncthreads();

    // prefetch next K-tile (global_prefetch_b8)
    if (k0 + BK < K) {
      int pr = m0 + ar; if (pr >= M) pr = M - 1;
      __builtin_prefetch(&A[(long long)pr * lda + k0 + BK + ac], 0, 1);
      int pn = n0 + bn; if (pn >= N) pn = N - 1;
      __builtin_prefetch(&Bw[(long long)pn * ldbK + k0 + BK + bk], 0, 1);
    }

    // ---------- fragments + WMMA ----------
    v16bf af[2], bfr[4];
#pragma unroll
    for (int i = 0; i < 2; ++i) {
      int m = wm * 32 + i * 16 + l16;
      af[i] = mk16(*(const bf16x8*)&As[m][half * 8],
                   *(const bf16x8*)&As[m][16 + half * 8]);
    }
#pragma unroll
    for (int j = 0; j < 4; ++j) {
      int n = wn * 64 + j * 16 + l16;
      bfr[j] = mk16(*(const bf16x8*)&Bs[n][half * 8],
                    *(const bf16x8*)&Bs[n][16 + half * 8]);
    }
#pragma unroll
    for (int i = 0; i < 2; ++i)
#pragma unroll
      for (int j = 0; j < 4; ++j)
        acc[i][j] = __builtin_amdgcn_wmma_f32_16x16x32_bf16(
            false, af[i], false, bfr[j], (short)0, acc[i][j], false, false);

    __syncthreads();
  }

  // ---------- epilogue ----------
#pragma unroll
  for (int i = 0; i < 2; ++i) {
#pragma unroll
    for (int j = 0; j < 4; ++j) {
      int nCol = n0 + wn * 64 + j * 16 + l16;
      if (nCol >= N) continue;
      float bv = biasp ? biasp[nCol] : 0.f;
#pragma unroll
      for (int r = 0; r < 8; ++r) {
        int mRow = m0 + wm * 32 + i * 16 + half * 8 + r;
        if (mRow >= M) continue;
        float v = acc[i][j][r] + bv;
        if (ACT == 1) v = softsign(v);
        int outRow = mRow;
        if (ACCUM == 1) {
          if (remapT > 0) {
            int bb = mRow / remapT, p = mRow % remapT;
            int tt = rev ? (remapT - 1 - p) : p;
            outRow = bb * remapT + tt;
          }
          C[(long long)outRow * ldc + nCol] += v;
        } else {
          C[(long long)outRow * ldc + nCol] = v;
        }
      }
    }
  }
}

// =====================================================================
// Helper kernels
// =====================================================================
// plain f32 -> bf16 conversion (layout preserved: weights are already [N][K])
__global__ void convert_bf16(const float* __restrict__ in, bf16_t* __restrict__ out,
                             long long total4) {   // total/4 vec4 groups
  long long idx = (long long)blockIdx.x * 256 + threadIdx.x;
  if (idx >= total4) return;
  float4u v = *(const float4u*)(in + idx * 4);
  bf16_t* o = out + idx * 4;
  o[0] = (bf16_t)v[0]; o[1] = (bf16_t)v[1]; o[2] = (bf16_t)v[2]; o[3] = (bf16_t)v[3];
}

// day_w is [day][K][N]; emit selected-day weights transposed to [N][K] bf16
__global__ void gather_day_w(const float* __restrict__ day_w, const int* __restrict__ day_idx,
                             bf16_t* __restrict__ Wsel) {
  int b = blockIdx.y;
  int e = blockIdx.x * 256 + threadIdx.x;        // 0..512*512-1, e = n*512 + k
  if (e >= NC * NC) return;
  int n = e >> 9, k = e & (NC - 1);
  Wsel[(long long)b * NC * NC + e] =
      (bf16_t)day_w[(long long)day_idx[b] * NC * NC + (long long)k * NC + n];
}

__global__ void gather_day_b(const float* __restrict__ day_b, const int* __restrict__ day_idx,
                             float* __restrict__ biasSel) {
  int b = blockIdx.y;
  int e = blockIdx.x * 256 + threadIdx.x;
  if (e < NC) biasSel[b * NC + e] = day_b[day_idx[b] * NC + e];
}

__global__ void copy_xt(const float* __restrict__ x, float* __restrict__ hfull) {
  int idx = blockIdx.x * 256 + threadIdx.x;      // b*T + t
  if (idx < BB * TT_IN)
    hfull[(long long)idx * D_IN + NC] = x[(long long)idx * D_IN + NC];
}

__global__ void patch_gather(const float* __restrict__ hfull, float* __restrict__ Apatch) {
  long long idx = (long long)blockIdx.x * 256 + threadIdx.x;
  if (idx >= (long long)ROWS * (PATCH * D_IN)) return;
  int r = (int)(idx / (PATCH * D_IN));
  int j = (int)(idx % (PATCH * D_IN));
  int b = r / T_OUT, p = r % T_OUT;
  int pi = j / D_IN, d = j % D_IN;
  Apatch[idx] = hfull[((long long)b * TT_IN + p * STRIDE_P + pi) * D_IN + d];
}

__global__ __launch_bounds__(256) void layernorm_rows(const float* __restrict__ in,
                                                      float* __restrict__ out,
                                                      const float* __restrict__ w,
                                                      const float* __restrict__ b, int rows) {
  int row = blockIdx.x * 8 + (threadIdx.x >> 5);
  int lane = threadIdx.x & 31;
  if (row >= rows) return;
  const float* xr = in + (long long)row * D_MODEL;
  float v[16], s = 0.f;
#pragma unroll
  for (int i = 0; i < 16; ++i) { v[i] = xr[lane + 32 * i]; s += v[i]; }
#pragma unroll
  for (int o = 16; o > 0; o >>= 1) s += __shfl_xor(s, o);
  float mu = s * (1.f / D_MODEL);
  float q = 0.f;
#pragma unroll
  for (int i = 0; i < 16; ++i) { float d = v[i] - mu; q += d * d; }
#pragma unroll
  for (int o = 16; o > 0; o >>= 1) q += __shfl_xor(q, o);
  float rs = rsqrtf(q * (1.f / D_MODEL) + EPSV);
  float* orow = out + (long long)row * D_MODEL;
#pragma unroll
  for (int i = 0; i < 16; ++i) {
    int c = lane + 32 * i;
    orow[c] = (v[i] - mu) * rs * w[c] + b[c];
  }
}

// depthwise causal conv (taps=4) + bias + SiLU; output written in scan order
__global__ void conv_silu(const float* __restrict__ zx, const float* __restrict__ cw,
                          const float* __restrict__ cb, float* __restrict__ convX, int rev) {
  long long idx = (long long)blockIdx.x * 256 + threadIdx.x;
  if (idx >= (long long)ROWS * CONVD) return;
  int c = (int)(idx % CONVD);
  int rp = (int)(idx / CONVD);
  int b = rp / T_OUT, p = rp % T_OUT;
  float acc = cb[c];
#pragma unroll
  for (int j = 0; j < DCONV; ++j) {
    int pp = p - (DCONV - 1) + j;
    if (pp < 0) continue;
    int tt = rev ? (T_OUT - 1 - pp) : pp;
    acc += cw[c * DCONV + j] * zx[(long long)(b * T_OUT + tt) * DINPROJ + DI + c];
  }
  convX[idx] = silu(acc);
}

// register-resident selective scan: one block per (batch, head)
__global__ __launch_bounds__(256) void scan_kernel(const float* __restrict__ zx,
                                                   const float* __restrict__ convX,
                                                   const float* __restrict__ dt_bias,
                                                   const float* __restrict__ A_log,
                                                   const float* __restrict__ Dp,
                                                   float* __restrict__ ybuf, int rev) {
  int b  = blockIdx.x >> 4;
  int hh = blockIdx.x & 15;
  int tid = threadIdx.x;
  int p  = tid >> 2;            // head-dim index 0..63
  int nb = (tid & 3) * 16;      // state-col base
  float st[16];
#pragma unroll
  for (int j = 0; j < 16; ++j) st[j] = 0.f;
  float Acoef = -expf(A_log[hh]);
  float db = dt_bias[hh];
  float Dh = Dp[hh];
  for (int step = 0; step < T_OUT; ++step) {
    int tt = rev ? (T_OUT - 1 - step) : step;
    long long rowz = (long long)(b * T_OUT + tt) * DINPROJ;
    long long rowc = (long long)(b * T_OUT + step) * CONVD;
    // prefetch next step's operands while this step computes
    if (step + 1 < T_OUT) {
      __builtin_prefetch(&convX[rowc + CONVD + hh * HEADDIM + p], 0, 1);
      __builtin_prefetch(&convX[rowc + CONVD + DI + nb], 0, 1);
    }
    float dtr = zx[rowz + DI + CONVD + hh] + db;
    float dt = (dtr > 20.f) ? dtr : log1pf(expf(dtr));   // softplus
    float dA = expf(dt * Acoef);
    float xv = convX[rowc + hh * HEADDIM + p];
    float px = dt * xv;
    const float* Bp = &convX[rowc + DI + nb];
    const float* Cp = &convX[rowc + DI + DSTATE + nb];
    float y = 0.f;
#pragma unroll
    for (int j = 0; j < 16; ++j) {
      st[j] = st[j] * dA + px * Bp[j];
      y += st[j] * Cp[j];
    }
    y += __shfl_xor(y, 1);
    y += __shfl_xor(y, 2);
    if ((tid & 3) == 0)
      ybuf[(long long)(b * T_OUT + step) * DI + hh * HEADDIM + p] = y + Dh * xv;
  }
}

// y = RMSNorm(y * silu(z)) * norm_w     (one block per row)
__global__ __launch_bounds__(256) void gate_rmsnorm(float* __restrict__ ybuf,
                                                    const float* __restrict__ zx,
                                                    const float* __restrict__ normw, int rev) {
  int row = blockIdx.x;
  int b = row / T_OUT, p = row % T_OUT;
  int tt = rev ? (T_OUT - 1 - p) : p;
  const float* z = zx + (long long)(b * T_OUT + tt) * DINPROJ;
  float* y = ybuf + (long long)row * DI;
  int tid = threadIdx.x;
  float vals[4], ss = 0.f;
#pragma unroll
  for (int i = 0; i < 4; ++i) {
    int c = tid + 256 * i;
    float v = y[c] * silu(z[c]);
    vals[i] = v; ss += v * v;
  }
#pragma unroll
  for (int o = 16; o > 0; o >>= 1) ss += __shfl_xor(ss, o);
  __shared__ float red[8];
  if ((tid & 31) == 0) red[tid >> 5] = ss;
  __syncthreads();
  float tot = 0.f;
#pragma unroll
  for (int i = 0; i < 8; ++i) tot += red[i];
  float rs = rsqrtf(tot * (1.f / DI) + EPSV);
#pragma unroll
  for (int i = 0; i < 4; ++i) {
    int c = tid + 256 * i;
    y[c] = vals[i] * rs * normw[c];
  }
}

__global__ void logits_kernel(const float* __restrict__ h, const float* __restrict__ ow,
                              const float* __restrict__ ob, float* __restrict__ out) {
  long long idx = (long long)blockIdx.x * 256 + threadIdx.x;
  if (idx >= (long long)ROWS * N_CLASSES) return;
  int r = (int)(idx / N_CLASSES), c = (int)(idx % N_CLASSES);
  const float* hv = h + (long long)r * D_MODEL;
  const float* wv = ow + (long long)c * D_MODEL;
  float s = ob[c];
  for (int k = 0; k < D_MODEL; ++k) s += hv[k] * wv[k];
  out[idx] = s;
}

// =====================================================================
// Host launcher
// =====================================================================
extern "C" void kernel_launch(void* const* d_in, const int* in_sizes, int n_in,
                              void* d_out, int out_size, void* d_ws, size_t ws_size,
                              hipStream_t stream) {
  (void)in_sizes; (void)n_in; (void)out_size; (void)ws_size;
  const float* x        = (const float*)d_in[0];
  const int*   day_idx  = (const int*)  d_in[1];
  const float* day_w    = (const float*)d_in[2];
  const float* day_b    = (const float*)d_in[3];
  const float* w1       = (const float*)d_in[4];
  const float* b1       = (const float*)d_in[5];
  const float* w2       = (const float*)d_in[6];
  const float* b2       = (const float*)d_in[7];
  const float* ln_w     = (const float*)d_in[8];
  const float* ln_b     = (const float*)d_in[9];
  const float* m_in_w   = (const float*)d_in[10];
  const float* m_conv_w = (const float*)d_in[11];
  const float* m_conv_b = (const float*)d_in[12];
  const float* m_dt     = (const float*)d_in[13];
  const float* m_Alog   = (const float*)d_in[14];
  const float* m_D      = (const float*)d_in[15];
  const float* m_norm_w = (const float*)d_in[16];
  const float* m_out_w  = (const float*)d_in[17];
  const float* out_w    = (const float*)d_in[18];
  const float* out_b    = (const float*)d_in[19];
  float* out = (float*)d_out;

  // ---- bump allocator on workspace ----
  char* ws = (char*)d_ws;
  size_t off = 0;
  auto alloc = [&](size_t bytes) -> void* {
    void* p = ws + off;
    off = (off + bytes + 255) & ~(size_t)255;
    return p;
  };
  const size_t KPATCH = (size_t)PATCH * D_IN;                 // 7182
  bf16_t* Wsel    = (bf16_t*)alloc((size_t)BB * NC * NC * 2);
  float*  biasSel = (float*) alloc((size_t)BB * NC * 4);
  bf16_t* w1t     = (bf16_t*)alloc((size_t)PROJ * KPATCH * 2);
  bf16_t* w2t     = (bf16_t*)alloc((size_t)D_MODEL * PROJ * 2);
  bf16_t* inWt    = (bf16_t*)alloc((size_t)NMAMBA * DINPROJ * D_MODEL * 2);
  bf16_t* outWt   = (bf16_t*)alloc((size_t)NMAMBA * D_MODEL * DI * 2);
  float*  hfull   = (float*) alloc((size_t)BB * TT_IN * D_IN * 4);
  float*  hbuf    = (float*) alloc((size_t)ROWS * D_MODEL * 4);
  // union region: {Apatch, P1} (stage 3) reused by {hn, zx, convX, ybuf} (layers)
  size_t szApatch = (size_t)ROWS * KPATCH * 4;
  size_t szP1     = (size_t)ROWS * PROJ * 4;
  size_t szHn     = (size_t)ROWS * D_MODEL * 4;
  size_t szZx     = (size_t)ROWS * DINPROJ * 4;
  size_t szConv   = (size_t)ROWS * CONVD * 4;
  size_t szY      = (size_t)ROWS * DI * 4;
  size_t a2 = ((szApatch + 255) & ~(size_t)255);
  size_t u1 = a2 + szP1;
  size_t h1 = ((szHn + 255) & ~(size_t)255), h2 = h1 + ((szZx + 255) & ~(size_t)255),
         h3 = h2 + ((szConv + 255) & ~(size_t)255), u2 = h3 + szY;
  char* U = (char*)alloc(u1 > u2 ? u1 : u2);
  float* Apatch = (float*)U;
  float* P1     = (float*)(U + a2);
  float* hn     = (float*)U;
  float* zx     = (float*)(U + h1);
  float* convX  = (float*)(U + h2);
  float* ybuf   = (float*)(U + h3);

  dim3 blk(256);

  // ---- 1. weight conversion (fp32 -> bf16, layouts already [N][K]) ----
  {
    long long t4;
    t4 = (long long)PROJ * KPATCH / 4;
    convert_bf16<<<dim3((unsigned)((t4 + 255) / 256)), blk, 0, stream>>>(w1, w1t, t4);
    t4 = (long long)D_MODEL * PROJ / 4;
    convert_bf16<<<dim3((unsigned)((t4 + 255) / 256)), blk, 0, stream>>>(w2, w2t, t4);
    t4 = (long long)NMAMBA * DINPROJ * D_MODEL / 4;
    convert_bf16<<<dim3((unsigned)((t4 + 255) / 256)), blk, 0, stream>>>(m_in_w, inWt, t4);
    t4 = (long long)NMAMBA * D_MODEL * DI / 4;
    convert_bf16<<<dim3((unsigned)((t4 + 255) / 256)), blk, 0, stream>>>(m_out_w, outWt, t4);
    gather_day_w<<<dim3(1024, BB), blk, 0, stream>>>(day_w, day_idx, Wsel);
    gather_day_b<<<dim3(2, BB), blk, 0, stream>>>(day_b, day_idx, biasSel);
  }

  // ---- 2. day-specific transform + softsign -> hfull[:, :, :512] ----
  wmma_gemm<1, 0><<<dim3(NC / BN, TT_IN / BM, BB), blk, 0, stream>>>(
      x, (long long)TT_IN * D_IN, D_IN,
      Wsel, (long long)NC * NC, NC,
      biasSel, NC,
      hfull, (long long)TT_IN * D_IN, D_IN,
      TT_IN, NC, NC, 0, 0);
  copy_xt<<<dim3((BB * TT_IN + 255) / 256), blk, 0, stream>>>(x, hfull);

  // ---- 3. patchify + two projections ----
  {
    long long t = (long long)ROWS * KPATCH;
    patch_gather<<<dim3((unsigned)((t + 255) / 256)), blk, 0, stream>>>(hfull, Apatch);
  }
  wmma_gemm<1, 0><<<dim3(PROJ / BN, (ROWS + BM - 1) / BM, 1), blk, 0, stream>>>(
      Apatch, 0, (int)KPATCH, w1t, 0, (int)KPATCH, b1, 0,
      P1, 0, PROJ, ROWS, PROJ, (int)KPATCH, 0, 0);
  wmma_gemm<0, 0><<<dim3(D_MODEL / BN, (ROWS + BM - 1) / BM, 1), blk, 0, stream>>>(
      P1, 0, PROJ, w2t, 0, PROJ, b2, 0,
      hbuf, 0, D_MODEL, ROWS, D_MODEL, PROJ, 0, 0);

  // ---- 4. bidirectional Mamba2 layers ----
  for (int layer = 0; layer < NLAYER; ++layer) {
    layernorm_rows<<<dim3((ROWS + 7) / 8), blk, 0, stream>>>(
        hbuf, hn, ln_w + layer * D_MODEL, ln_b + layer * D_MODEL, ROWS);
    for (int dir = 0; dir < 2; ++dir) {
      int mi = 2 * layer + dir;
      int rev = dir;
      // in-projection (per-timestep; flip commutes)
      wmma_gemm<0, 0><<<dim3((DINPROJ + BN - 1) / BN, (ROWS + BM - 1) / BM, 1), blk, 0, stream>>>(
          hn, 0, D_MODEL, inWt + (size_t)mi * DINPROJ * D_MODEL, 0, D_MODEL,
          nullptr, 0, zx, 0, DINPROJ, ROWS, DINPROJ, D_MODEL, 0, 0);
      // causal depthwise conv + SiLU (in scan order)
      {
        long long t = (long long)ROWS * CONVD;
        conv_silu<<<dim3((unsigned)((t + 255) / 256)), blk, 0, stream>>>(
            zx, m_conv_w + (size_t)mi * CONVD * DCONV, m_conv_b + (size_t)mi * CONVD, convX, rev);
      }
      // selective scan (register-resident state)
      scan_kernel<<<dim3(BB * NH), blk, 0, stream>>>(
          zx, convX, m_dt + mi * NH, m_Alog + mi * NH, m_D + mi * NH, ybuf, rev);
      // gate + RMSNorm
      gate_rmsnorm<<<dim3(ROWS), blk, 0, stream>>>(ybuf, zx, m_norm_w + (size_t)mi * DI, rev);
      // out-projection, accumulate into h with time-reverse remap for backward dir
      wmma_gemm<0, 1><<<dim3(D_MODEL / BN, (ROWS + BM - 1) / BM, 1), blk, 0, stream>>>(
          ybuf, 0, DI, outWt + (size_t)mi * D_MODEL * DI, 0, DI,
          nullptr, 0, hbuf, 0, D_MODEL, ROWS, D_MODEL, DI, T_OUT, rev);
    }
  }

  // ---- 5. classifier head ----
  {
    long long t = (long long)ROWS * N_CLASSES;
    logits_kernel<<<dim3((unsigned)((t + 255) / 256)), blk, 0, stream>>>(hbuf, out_w, out_b, out);
  }
}
